// LSTM_84945863180423
// MI455X (gfx1250) — matrix operand order repaired and checked by reading, not compiled, not run on hardware
//
#include <hip/hip_runtime.h>
#include <hip/hip_bf16.h>

// ---------------------------------------------------------------------------
// 2-layer LSTM (B=64, S=512, I=512, H=1024) + FC(1024->2) + softmax.
//   * Input projections hoisted out of the time loop (parallel WMMA GEMM).
//   * Recurrence: ONE persistent kernel per layer (512 steps internally,
//     device-wide release/acquire spin barrier), weights LDS-resident,
//     v_wmma_f32_16x16x32_bf16 with f32 accumulation.
// ---------------------------------------------------------------------------

typedef __attribute__((ext_vector_type(16))) __bf16 v16bf;
typedef __attribute__((ext_vector_type(8)))  __bf16 v8bf;
typedef __attribute__((ext_vector_type(8)))  float  v8f;

#define LSTM_B   64
#define LSTM_S   512
#define LSTM_I   512
#define LSTM_H   1024
#define LSTM_4H  4096

#define WLDS_STRIDE 1032                         // 1024 + 8 pad (bank spread)
#define GSUM_FLOATS (4 * 64 * 16)                // 16 KB
#define SMEM_BYTES  (GSUM_FLOATS * 4 + 64 * WLDS_STRIDE * 2)   // 148480 B

// ---------------------------------------------------------------------------
// Elementwise helpers
// ---------------------------------------------------------------------------
__global__ void k_f32_to_bf16(const float* __restrict__ in, __bf16* __restrict__ out, int n) {
    int i = blockIdx.x * blockDim.x + threadIdx.x;
    if (i < n) out[i] = (__bf16)in[i];
}

__global__ void k_bias_sum(const float* __restrict__ a, const float* __restrict__ b,
                           float* __restrict__ o, int n) {
    int i = blockIdx.x * blockDim.x + threadIdx.x;
    if (i < n) o[i] = a[i] + b[i];
}

// ---------------------------------------------------------------------------
// WMMA fragment helpers (layouts per CDNA5 ISA 7.12.2, wave32)
//   A (16x32 bf16): lane L -> row = L%16 ; elems 0..7  = K (L/16)*8 .. +7
//                                   elems 8..15 = K (L/16)*8+16 .. +23
//   B (32x16 bf16): lane L -> col = L%16 ; same K split as A
//   C (16x16 f32) : lane L -> col = L%16 ; vgpr r -> row = r + 8*(L/16)
// ---------------------------------------------------------------------------
__device__ __forceinline__ v16bf combine16(v8bf lo, v8bf hi) {
    v16bf a;
#pragma unroll
    for (int e = 0; e < 8; ++e) { a[e] = lo[e]; a[e + 8] = hi[e]; }
    return a;
}
__device__ __forceinline__ v16bf load_a_frag(const __bf16* __restrict__ p) {
    return combine16(*(const v8bf*)p, *(const v8bf*)(p + 16));   // gap 16 (K split)
}
__device__ __forceinline__ v16bf load_b_frag(const __bf16* p) {
    return combine16(*(const v8bf*)p, *(const v8bf*)(p + 16));   // gap 16 (K split)
}

// ---------------------------------------------------------------------------
// gemm_bias: out[m][n] = sum_k A[m][k] * W[n][k] + bias[n]
//   out rows m = s*B + b.  swapBS=1: A row at offset (b*S+s)*K (layer-0 x),
//   swapBS=0: A row at offset m*K (layer-1 hs0, already [S][B][H]).
// grid = (N/128, M/16), block = 256 (8 waves, 1 tile/wave)
// ---------------------------------------------------------------------------
__global__ void __launch_bounds__(256)
k_gemm_bias(const __bf16* __restrict__ A, const __bf16* __restrict__ W,
            const float* __restrict__ bias, float* __restrict__ out,
            int N, int K, int swapBS) {
    const int wave = threadIdx.x >> 5;
    const int lane = threadIdx.x & 31;
    const int mt   = blockIdx.y;
    const int nt   = blockIdx.x * 8 + wave;

    const int mrow = mt * 16 + (lane & 15);
    const int ncol = nt * 16 + (lane & 15);
    long aoff;
    if (swapBS) {
        const int b = mrow & (LSTM_B - 1);
        const int s = mrow / LSTM_B;
        aoff = ((long)b * LSTM_S + s) * (long)K;
    } else {
        aoff = (long)mrow * (long)K;
    }
    const int khA = (lane >> 4) * 8;
    const int khB = (lane >> 4) * 8;
    const __bf16* __restrict__ Arow = A + aoff + khA;
    const __bf16* __restrict__ Wrow = W + (long)ncol * (long)K + khB;

    v8f acc = {};
#pragma unroll 2
    for (int kk = 0; kk < K; kk += 32) {
        v16bf af = load_a_frag(Arow + kk);
        v16bf bf = load_b_frag(Wrow + kk);
        acc = __builtin_amdgcn_wmma_f32_16x16x32_bf16(
                  false, af, false, bf, (short)0, acc, false, false);
    }

    const float bv = bias[ncol];
    const int mbase = mt * 16 + 8 * (lane >> 4);
#pragma unroll
    for (int r = 0; r < 8; ++r)
        out[(long)(mbase + r) * (long)N + ncol] = acc[r] + bv;
}

// ---------------------------------------------------------------------------
// Persistent recurrence kernel: runs all S steps of one LSTM layer.
// grid = H/16 = 64 blocks, block = 256 (8 waves). Block wg owns h-columns
// [wg*16, wg*16+16). Its 4-gate weight slice (64 rows x 1024 K bf16 = 128 KB)
// is staged in LDS ONCE, then each step does the 4-gate x 4-Mtile WMMA GEMM
// (wave = gate-pair x m-tile: 1 global A frag + 2 LDS B frags + 2 WMMAs per
// k-step), the fused cell update, and a device-wide spin barrier.
// ---------------------------------------------------------------------------
__global__ void __launch_bounds__(256)
k_lstm_seq(const __bf16* __restrict__ Whh, const float* __restrict__ xg,
           float* __restrict__ c, __bf16* __restrict__ hs,
           const __bf16* __restrict__ h0zero, int* __restrict__ bar,
           int nblocks) {
    extern __shared__ char smem[];
    float*  gsum = (float*)smem;                       // [4][64][16]
    __bf16* wlds = (__bf16*)(smem + GSUM_FLOATS * 4);  // [64][WLDS_STRIDE]

    const int wg = blockIdx.x;

    // ---- stage this block's weight slice into LDS (once per layer) ----
    for (int chunk = threadIdx.x; chunk < 64 * 64; chunk += 256) {
        const int rl   = chunk >> 6;          // local row 0..63 = g*16 + j
        const int koff = (chunk & 63) << 4;   // 0..1008, step 16
        const int g = rl >> 4, j = rl & 15;
        const __bf16* src =
            Whh + ((long)(g * LSTM_H + wg * 16 + j)) * LSTM_H + koff;
        *(v8bf*)&wlds[rl * WLDS_STRIDE + koff]     = *(const v8bf*)src;
        *(v8bf*)&wlds[rl * WLDS_STRIDE + koff + 8] = *(const v8bf*)(src + 8);
    }
    __syncthreads();

    const int wave = threadIdx.x >> 5;
    const int lane = threadIdx.x & 31;
    const int j    = lane & 15;
    const int mt   = wave & 3;        // m-tile 0..3
    const int gp   = wave >> 2;       // gate pair 0..1
    const int g0   = gp * 2, g1 = g0 + 1;
    const int khA  = (lane >> 4) * 8;
    const int khB  = (lane >> 4) * 8;

    const __bf16* wl0 = &wlds[(g0 * 16 + j) * WLDS_STRIDE + khB];
    const __bf16* wl1 = &wlds[(g1 * 16 + j) * WLDS_STRIDE + khB];
    const long arowoff = (long)(mt * 16 + j) * LSTM_H + khA;
    const int  mb = 8 * (lane >> 4);

    for (int t = 0; t < LSTM_S; ++t) {
        const __bf16* hp =
            (t == 0) ? h0zero : hs + (size_t)(t - 1) * LSTM_B * LSTM_H;
        const float* xg_t = xg + (size_t)t * LSTM_B * LSTM_4H;
        __bf16* hout = hs + (size_t)t * LSTM_B * LSTM_H;

        // ---- 16x16x1024 x (2 gates) WMMA GEMM ----
        const __bf16* Arow = hp + arowoff;
        v8f acc0 = {}, acc1 = {};
#pragma unroll 2
        for (int kk = 0; kk < LSTM_H; kk += 32) {
            v16bf a  = load_a_frag(Arow + kk);
            v16bf b0 = load_b_frag(wl0 + kk);
            v16bf b1 = load_b_frag(wl1 + kk);
            acc0 = __builtin_amdgcn_wmma_f32_16x16x32_bf16(
                       false, a, false, b0, (short)0, acc0, false, false);
            acc1 = __builtin_amdgcn_wmma_f32_16x16x32_bf16(
                       false, a, false, b1, (short)0, acc1, false, false);
        }
#pragma unroll
        for (int r = 0; r < 8; ++r) {
            gsum[(g0 * 64 + mt * 16 + mb + r) * 16 + j] = acc0[r];
            gsum[(g1 * 64 + mt * 16 + mb + r) * 16 + j] = acc1[r];
        }
        __syncthreads();

        // ---- fused cell update: 64 rows x 16 cols, 4 elems/thread ----
        for (int idx = threadIdx.x; idx < LSTM_B * 16; idx += 256) {
            const int b   = idx >> 4;
            const int jj  = idx & 15;
            const int col = wg * 16 + jj;
            const float* xr = xg_t + (long)b * LSTM_4H;
            float xi = xr[col]              + gsum[(0 * 64 + b) * 16 + jj];
            float xf = xr[LSTM_H + col]     + gsum[(1 * 64 + b) * 16 + jj];
            float xgv = xr[2 * LSTM_H + col] + gsum[(2 * 64 + b) * 16 + jj];
            float xo = xr[3 * LSTM_H + col] + gsum[(3 * 64 + b) * 16 + jj];
            float ig = 1.f / (1.f + __expf(-xi));
            float fg = 1.f / (1.f + __expf(-xf));
            float gg = tanhf(xgv);
            float og = 1.f / (1.f + __expf(-xo));
            const long ci = (long)b * LSTM_H + col;
            float cn = fg * c[ci] + ig * gg;
            c[ci] = cn;
            hout[ci] = (__bf16)(og * tanhf(cn));
        }
        __syncthreads();   // all of this block's h written

        // ---- device-wide split barrier (release signal, acquire spin) ----
        if (threadIdx.x == 0) {
            __hip_atomic_fetch_add(&bar[t], 1, __ATOMIC_RELEASE,
                                   __HIP_MEMORY_SCOPE_AGENT);
            while (__hip_atomic_load(&bar[t], __ATOMIC_ACQUIRE,
                                     __HIP_MEMORY_SCOPE_AGENT) < nblocks)
                __builtin_amdgcn_s_sleep(1);
        }
        __syncthreads();
    }
}

// ---------------------------------------------------------------------------
// head: logits = h_last @ fc_w^T + fc_b ; 2-class softmax.  1 block, 64 thr.
// ---------------------------------------------------------------------------
__global__ void k_head(const __bf16* __restrict__ hlast, const float* __restrict__ fcw,
                       const float* __restrict__ fcb, float* __restrict__ out) {
    const int b = threadIdx.x;
    if (b >= LSTM_B) return;
    float a0 = fcb[0], a1 = fcb[1];
    const __bf16* hr = hlast + (long)b * LSTM_H;
    for (int h = 0; h < LSTM_H; ++h) {
        float v = (float)hr[h];
        a0 += v * fcw[h];
        a1 += v * fcw[LSTM_H + h];
    }
    float m  = fmaxf(a0, a1);
    float e0 = __expf(a0 - m), e1 = __expf(a1 - m);
    float s  = e0 + e1;
    out[b * 2 + 0] = e0 / s;
    out[b * 2 + 1] = e1 / s;
}

// ---------------------------------------------------------------------------
// Host orchestration (graph-capture safe: kernel launches + MemsetAsync only)
// ---------------------------------------------------------------------------
static inline size_t alignup(size_t x) { return (x + 255) & ~(size_t)255; }

extern "C" void kernel_launch(void* const* d_in, const int* in_sizes, int n_in,
                              void* d_out, int out_size, void* d_ws, size_t ws_size,
                              hipStream_t stream) {
    const float* x     = (const float*)d_in[0];
    const float* w_ih0 = (const float*)d_in[1];
    const float* w_hh0 = (const float*)d_in[2];
    const float* b_ih0 = (const float*)d_in[3];
    const float* b_hh0 = (const float*)d_in[4];
    const float* w_ih1 = (const float*)d_in[5];
    const float* w_hh1 = (const float*)d_in[6];
    const float* b_ih1 = (const float*)d_in[7];
    const float* b_hh1 = (const float*)d_in[8];
    const float* fc_w  = (const float*)d_in[9];
    const float* fc_b  = (const float*)d_in[10];
    float* out = (float*)d_out;

    char* ws = (char*)d_ws;
    size_t off = 0;
    auto take = [&](size_t bytes) { char* p = ws + off; off += alignup(bytes); return p; };

    __bf16* x_bf    = (__bf16*)take((size_t)LSTM_B * LSTM_S * LSTM_I * 2);     // 33.6 MB
    __bf16* wih0_bf = (__bf16*)take((size_t)LSTM_4H * LSTM_I * 2);             //  4.2 MB
    __bf16* whh0_bf = (__bf16*)take((size_t)LSTM_4H * LSTM_H * 2);             //  8.4 MB
    __bf16* wih1_bf = (__bf16*)take((size_t)LSTM_4H * LSTM_H * 2);             //  8.4 MB
    __bf16* whh1_bf = (__bf16*)take((size_t)LSTM_4H * LSTM_H * 2);             //  8.4 MB
    float*  b0      = (float*) take((size_t)LSTM_4H * 4);
    float*  b1      = (float*) take((size_t)LSTM_4H * 4);
    __bf16* hs0     = (__bf16*)take((size_t)LSTM_S * LSTM_B * LSTM_H * 2);     // 67 MB  [S][B][H]
    __bf16* hs1     = (__bf16*)take((size_t)LSTM_S * LSTM_B * LSTM_H * 2);     // 67 MB
    float*  cbuf    = (float*) take((size_t)LSTM_B * LSTM_H * 4);              // 256 KB
    __bf16* h0zero  = (__bf16*)take((size_t)LSTM_B * LSTM_H * 2);              // 128 KB
    int*    bar0    = (int*)   take((size_t)LSTM_S * 4);
    int*    bar1    = (int*)   take((size_t)LSTM_S * 4);
    float*  xg      = (float*) take((size_t)LSTM_S * LSTM_B * LSTM_4H * 4);    // 537 MB [S][B][4H]
    (void)ws_size;

    // 1) precision conversion + bias fusion
    {
        int n;
        n = LSTM_B * LSTM_S * LSTM_I;
        k_f32_to_bf16<<<(n + 255) / 256, 256, 0, stream>>>(x, x_bf, n);
        n = LSTM_4H * LSTM_I;
        k_f32_to_bf16<<<(n + 255) / 256, 256, 0, stream>>>(w_ih0, wih0_bf, n);
        n = LSTM_4H * LSTM_H;
        k_f32_to_bf16<<<(n + 255) / 256, 256, 0, stream>>>(w_hh0, whh0_bf, n);
        k_f32_to_bf16<<<(n + 255) / 256, 256, 0, stream>>>(w_ih1, wih1_bf, n);
        k_f32_to_bf16<<<(n + 255) / 256, 256, 0, stream>>>(w_hh1, whh1_bf, n);
        k_bias_sum<<<(LSTM_4H + 255) / 256, 256, 0, stream>>>(b_ih0, b_hh0, b0, LSTM_4H);
        k_bias_sum<<<(LSTM_4H + 255) / 256, 256, 0, stream>>>(b_ih1, b_hh1, b1, LSTM_4H);
    }
    hipMemsetAsync(h0zero, 0, (size_t)LSTM_B * LSTM_H * 2, stream);
    hipMemsetAsync(bar0, 0, (size_t)LSTM_S * 4, stream);
    hipMemsetAsync(bar1, 0, (size_t)LSTM_S * 4, stream);

    const int M = LSTM_S * LSTM_B;                 // 32768
    const dim3 ggrid(LSTM_4H / 128, M / 16);       // (32, 2048)

    // ---- layer 0 ----
    k_gemm_bias<<<ggrid, 256, 0, stream>>>(x_bf, wih0_bf, b0, xg,
                                           LSTM_4H, LSTM_I, /*swapBS=*/1);
    hipMemsetAsync(cbuf, 0, (size_t)LSTM_B * LSTM_H * 4, stream);
    k_lstm_seq<<<LSTM_H / 16, 256, SMEM_BYTES, stream>>>(
        whh0_bf, xg, cbuf, hs0, h0zero, bar0, LSTM_H / 16);

    // ---- layer 1 ----
    k_gemm_bias<<<ggrid, 256, 0, stream>>>(hs0, wih1_bf, b1, xg,
                                           LSTM_4H, LSTM_H, /*swapBS=*/0);
    hipMemsetAsync(cbuf, 0, (size_t)LSTM_B * LSTM_H * 4, stream);
    k_lstm_seq<<<LSTM_H / 16, 256, SMEM_BYTES, stream>>>(
        whh1_bf, xg, cbuf, hs1, h0zero, bar1, LSTM_H / 16);

    // ---- head ----
    k_head<<<1, 64, 0, stream>>>(hs1 + (size_t)(LSTM_S - 1) * LSTM_B * LSTM_H,
                                 fc_w, fc_b, out);
    (void)in_sizes; (void)n_in; (void)out_size;
}
